// GAT_23261542875427
// MI455X (gfx1250) — matrix operand order repaired and checked by reading, not compiled, not run on hardware
//
#include <hip/hip_runtime.h>

typedef long long i64;
typedef _Float16 half_t;
typedef _Float16 v16h __attribute__((ext_vector_type(16)));
typedef float v8f __attribute__((ext_vector_type(8)));

#define NEG_SLOPE 0.2f

// ---------------------------------------------------------------------------
// Fragment loaders (CDNA5 ISA 7.12.2 layouts, f32 memory -> f16 fragments)
//   A (16x32): lanes 0-15 -> M=lane, K in {0..7,16..23};
//              lanes 16-31 -> M=lane-16, K in {8..15,24..31}
//   B (32x16): lane -> N=lane&15, K in {0..7,16..23} (lanes 0-15), {8..15,24..31} (lanes 16-31)
// ---------------------------------------------------------------------------
__device__ __forceinline__ v16h load_a_frag(const float* __restrict__ xrow,
                                            int k0, int hi) {
  const float4* p0 = (const float4*)(xrow + k0 + hi * 8);
  const float4* p1 = (const float4*)(xrow + k0 + hi * 8 + 16);
  float4 x0 = p0[0], x1 = p0[1], x2 = p1[0], x3 = p1[1];
  v16h a;
  a[0]=(half_t)x0.x;  a[1]=(half_t)x0.y;  a[2]=(half_t)x0.z;  a[3]=(half_t)x0.w;
  a[4]=(half_t)x1.x;  a[5]=(half_t)x1.y;  a[6]=(half_t)x1.z;  a[7]=(half_t)x1.w;
  a[8]=(half_t)x2.x;  a[9]=(half_t)x2.y;  a[10]=(half_t)x2.z; a[11]=(half_t)x2.w;
  a[12]=(half_t)x3.x; a[13]=(half_t)x3.y; a[14]=(half_t)x3.z; a[15]=(half_t)x3.w;
  return a;
}

__device__ __forceinline__ v16h load_b_frag(const float* __restrict__ wrow,
                                            int k0, int hi) {
  const float4* q = (const float4*)(wrow + k0 + hi * 8);
  const float4* q1 = (const float4*)(wrow + k0 + hi * 8 + 16);
  float4 w0 = q[0], w1 = q[1], w2 = q1[0], w3 = q1[1];
  v16h b;
  b[0]=(half_t)w0.x;  b[1]=(half_t)w0.y;  b[2]=(half_t)w0.z;  b[3]=(half_t)w0.w;
  b[4]=(half_t)w1.x;  b[5]=(half_t)w1.y;  b[6]=(half_t)w1.z;  b[7]=(half_t)w1.w;
  b[8]=(half_t)w2.x;  b[9]=(half_t)w2.y;  b[10]=(half_t)w2.z; b[11]=(half_t)w2.w;
  b[12]=(half_t)w3.x; b[13]=(half_t)w3.y; b[14]=(half_t)w3.z; b[15]=(half_t)w3.w;
  return b;
}

// ---------------------------------------------------------------------------
// WMMA GEMM: OUT[M,Nout] = X[M,KIN] @ W[Nout,KIN]^T
// One wave computes a 16(M) x 64(N) tile: one A fragment feeds 4 WMMAs per
// 32-wide K step (4x less X traffic than 16x16 tiling). KIN compile-time ->
// fully unrolled k-loop (KIN/32 * 4 v_wmma per kernel).
// ---------------------------------------------------------------------------
template <int KIN>
__global__ __launch_bounds__(32) void gemm_xwt_wmma(
    const float* __restrict__ X, const float* __restrict__ W,
    float* __restrict__ OUT, int M, int Nout)
{
  const int lane = threadIdx.x;
  const int r    = lane & 15;
  const int hi   = lane >> 4;
  const int m0   = blockIdx.x * 16;
  const int n0   = blockIdx.y * 64;

  int m  = m0 + r;
  int mc = (m < M) ? m : (M - 1);               // clamp loads; stores guarded
  const float* __restrict__ xrow = X + (size_t)mc * KIN;
  const float* __restrict__ wr0  = W + (size_t)(n0 + r) * KIN;       // n-subtile 0
  const float* __restrict__ wr1  = wr0 + (size_t)16 * KIN;           // n-subtile 1
  const float* __restrict__ wr2  = wr0 + (size_t)32 * KIN;           // n-subtile 2
  const float* __restrict__ wr3  = wr0 + (size_t)48 * KIN;           // n-subtile 3

  v8f acc0 = {}, acc1 = {}, acc2 = {}, acc3 = {};
#pragma unroll
  for (int k0 = 0; k0 < KIN; k0 += 32) {
    v16h a  = load_a_frag(xrow, k0, hi);
    v16h b0 = load_b_frag(wr0, k0, hi);
    acc0 = __builtin_amdgcn_wmma_f32_16x16x32_f16(false, a, false, b0, (short)0, acc0, false, false);
    v16h b1 = load_b_frag(wr1, k0, hi);
    acc1 = __builtin_amdgcn_wmma_f32_16x16x32_f16(false, a, false, b1, (short)0, acc1, false, false);
    v16h b2 = load_b_frag(wr2, k0, hi);
    acc2 = __builtin_amdgcn_wmma_f32_16x16x32_f16(false, a, false, b2, (short)0, acc2, false, false);
    v16h b3 = load_b_frag(wr3, k0, hi);
    acc3 = __builtin_amdgcn_wmma_f32_16x16x32_f16(false, a, false, b3, (short)0, acc3, false, false);
  }

  // C/D layout: elem e -> row m0 + e + 8*hi, col = lane&15 within each subtile
  float* outbase = OUT + (size_t)(m0 + hi * 8) * Nout + n0 + r;
  if (m0 + 16 <= M) {                            // full tile: unconditional stores
#pragma unroll
    for (int e = 0; e < 8; ++e) {
      float* p = outbase + (size_t)e * Nout;
      p[0]  = acc0[e];
      p[16] = acc1[e];
      p[32] = acc2[e];
      p[48] = acc3[e];
    }
  } else {
#pragma unroll
    for (int e = 0; e < 8; ++e) {
      if (m0 + e + hi * 8 < M) {
        float* p = outbase + (size_t)e * Nout;
        p[0]  = acc0[e];
        p[16] = acc1[e];
        p[32] = acc2[e];
        p[48] = acc3[e];
      }
    }
  }
}

// ---------------------------------------------------------------------------
// Helpers
// ---------------------------------------------------------------------------
__device__ __forceinline__ float lrelu(float v) {
  return v > 0.0f ? v : NEG_SLOPE * v;
}
// Monotone float <-> uint encoding for atomicMax over floats
__device__ __forceinline__ unsigned enc_key(float f) {
  unsigned u = __float_as_uint(f);
  return (u & 0x80000000u) ? ~u : (u | 0x80000000u);
}
__device__ __forceinline__ float dec_key(unsigned k) {
  return (k & 0x80000000u) ? __uint_as_float(k & 0x7fffffffu)
                           : __uint_as_float(~k);
}

__global__ void fill_f32(float* __restrict__ p, float v, long n) {
  long i = (long)blockIdx.x * blockDim.x + threadIdx.x;
  if (i < n) p[i] = v;
}
__global__ void fill_u32(unsigned* __restrict__ p, unsigned v, long n) {
  long i = (long)blockIdx.x * blockDim.x + threadIdx.x;
  if (i < n) p[i] = v;
}

// Per-node attention logits: asrc[n,h] = sum_c h[n,h,c]*a_src[h,c] (same for dst)
__global__ void node_alpha(const float* __restrict__ h,
                           const float* __restrict__ a_src,
                           const float* __restrict__ a_dst,
                           float* __restrict__ asrc, float* __restrict__ adst,
                           int N, int H, int C) {
  long i = (long)blockIdx.x * blockDim.x + threadIdx.x;
  if (i >= (long)N * H) return;
  int n = (int)(i / H), hh = (int)(i % H);
  const float4* hp = (const float4*)(h + (size_t)n * H * C + (size_t)hh * C);
  const float4* as = (const float4*)(a_src + (size_t)hh * C);
  const float4* ad = (const float4*)(a_dst + (size_t)hh * C);
  float s1 = 0.0f, s2 = 0.0f;
  for (int c = 0; c < C / 4; ++c) {
    float4 v = hp[c], a = as[c], d = ad[c];
    s1 += v.x * a.x + v.y * a.y + v.z * a.z + v.w * a.w;
    s2 += v.x * d.x + v.y * d.y + v.z * d.z + v.w * d.w;
  }
  asrc[i] = s1; adst[i] = s2;
}

// Segment max of leaky_relu(asrc[src]+adst[dst]) over incoming edges of dst
__global__ void edge_max(const i64* __restrict__ srcI, const i64* __restrict__ dstI,
                         int E, int N, const float* __restrict__ asrc,
                         const float* __restrict__ adst,
                         unsigned* __restrict__ mkey, int H) {
  int e = blockIdx.x * blockDim.x + threadIdx.x;
  if (e >= E + N) return;
  int s, d;
  if (e < E) { s = (int)srcI[e]; d = (int)dstI[e]; } else { s = d = e - E; }
  for (int h = 0; h < H; ++h) {
    float v = lrelu(asrc[(size_t)s * H + h] + adst[(size_t)d * H + h]);
    atomicMax(&mkey[(size_t)d * H + h], enc_key(v));
  }
}

// ex = exp(e - max); store per-edge, accumulate denominator per dst
__global__ void edge_exp(const i64* __restrict__ srcI, const i64* __restrict__ dstI,
                         int E, int N, const float* __restrict__ asrc,
                         const float* __restrict__ adst,
                         const unsigned* __restrict__ mkey,
                         float* __restrict__ den, float* __restrict__ eal, int H) {
  int e = blockIdx.x * blockDim.x + threadIdx.x;
  if (e >= E + N) return;
  int s, d;
  if (e < E) { s = (int)srcI[e]; d = (int)dstI[e]; } else { s = d = e - E; }
  for (int h = 0; h < H; ++h) {
    float v  = lrelu(asrc[(size_t)s * H + h] + adst[(size_t)d * H + h]);
    float m  = dec_key(mkey[(size_t)d * H + h]);
    float ex = __expf(v - m);
    eal[(size_t)e * H + h] = ex;
    atomicAdd(&den[(size_t)d * H + h], ex);
  }
}

// eal[e,h] /= den[dst(e),h]  -> final softmax alpha per edge
__global__ void edge_norm(const i64* __restrict__ dstI, int E, int N,
                          const float* __restrict__ den, float* __restrict__ eal,
                          int H) {
  long i = (long)blockIdx.x * blockDim.x + threadIdx.x;
  if (i >= (long)(E + N) * H) return;
  long e = i / H; int h = (int)(i % H);
  int d = (e < E) ? (int)dstI[e] : (int)(e - E);
  eal[i] = eal[i] / den[(size_t)d * H + h];
}

// out[dst, h, c] += h[src, h, c] * alpha[e, h]
// 256-thread blocks; each block covers 256/HC edges; hcShift = log2(H*C), cShift = log2(C)
__global__ void edge_msg(const i64* __restrict__ srcI, const i64* __restrict__ dstI,
                         int E, int N, const float* __restrict__ hfeat,
                         const float* __restrict__ eal, float* __restrict__ out,
                         int H, int hcShift, int cShift) {
  const int HC = 1 << hcShift;
  int f     = threadIdx.x & (HC - 1);
  int local = threadIdx.x >> hcShift;
  int e     = blockIdx.x * (256 >> hcShift) + local;
  if (e >= E + N) return;
  int s, d;
  if (e < E) { s = (int)srcI[e]; d = (int)dstI[e]; } else { s = d = e - E; }
  int hh = f >> cShift;
  float a = eal[(size_t)e * H + hh];
  atomicAdd(&out[(size_t)d * HC + f], hfeat[(size_t)s * HC + f] * a);
}

// out += bias, optional ELU
__global__ void bias_act(float* __restrict__ out, const float* __restrict__ b,
                         int N, int HC, int do_elu) {
  long i = (long)blockIdx.x * blockDim.x + threadIdx.x;
  if (i >= (long)N * HC) return;
  float v = out[i] + b[i % HC];
  if (do_elu) v = v > 0.0f ? v : (expf(v) - 1.0f);
  out[i] = v;
}

// Global mean-pool accumulation
__global__ void pool_sum(const float* __restrict__ h, const i64* __restrict__ batch,
                         float* __restrict__ pool, float* __restrict__ cnt,
                         int N, int C) {
  long i = (long)blockIdx.x * blockDim.x + threadIdx.x;
  if (i >= (long)N * C) return;
  int n = (int)(i / C), c = (int)(i % C);
  int g = (int)batch[n];
  atomicAdd(&pool[(size_t)g * C + c], h[(size_t)n * C + c]);
  if (c == 0) atomicAdd(&cnt[g], 1.0f);
}

// out[g,k] = (pool[g,:]/max(cnt,1)) . Wfc[k,:] + bfc[k]
__global__ void head_fc(const float* __restrict__ pool, const float* __restrict__ cnt,
                        const float* __restrict__ Wfc, const float* __restrict__ bfc,
                        float* __restrict__ out, int G, int C, int K) {
  int i = blockIdx.x * blockDim.x + threadIdx.x;
  if (i >= G * K) return;
  int g = i / K, k = i % K;
  float cc = fmaxf(cnt[g], 1.0f);
  float s = 0.0f;
  for (int c = 0; c < C; ++c) s += (pool[(size_t)g * C + c] / cc) * Wfc[(size_t)k * C + c];
  out[i] = s + bfc[k];
}

// ---------------------------------------------------------------------------
extern "C" void kernel_launch(void* const* d_in, const int* in_sizes, int n_in,
                              void* d_out, int out_size, void* d_ws, size_t ws_size,
                              hipStream_t stream) {
  (void)n_in; (void)out_size; (void)ws_size;

  const float* x    = (const float*)d_in[0];
  const i64*   edge = (const i64*)d_in[1];
  const i64*   bat  = (const i64*)d_in[2];
  const float* W1   = (const float*)d_in[3];
  const float* a1s  = (const float*)d_in[4];
  const float* a1d  = (const float*)d_in[5];
  const float* b1   = (const float*)d_in[6];
  const float* W2   = (const float*)d_in[7];
  const float* a2s  = (const float*)d_in[8];
  const float* a2d  = (const float*)d_in[9];
  const float* b2   = (const float*)d_in[10];
  const float* W3   = (const float*)d_in[11];
  const float* a3s  = (const float*)d_in[12];
  const float* a3d  = (const float*)d_in[13];
  const float* b3   = (const float*)d_in[14];
  const float* Wfc  = (const float*)d_in[15];
  const float* bfc  = (const float*)d_in[16];

  const int N  = in_sizes[2];          // batch vector has N entries
  const int E  = in_sizes[1] / 2;
  const int ET = E + N;                // edges + self-loops
  const int G  = 64, NCLS = 2;
  const i64* srcI = edge;
  const i64* dstI = edge + E;

  // ---- workspace carve-out (floats) ----
  float* ws = (float*)d_ws;
  size_t o = 0;
  float*    hbuf = ws + o; o += (size_t)N * 256;   // projected features
  float*    agg  = ws + o; o += (size_t)N * 256;   // aggregated output (ping-pong)
  float*    asrc = ws + o; o += (size_t)N * 4;
  float*    adst = ws + o; o += (size_t)N * 4;
  unsigned* mkey = (unsigned*)(ws + o); o += (size_t)N * 4;
  float*    den  = ws + o; o += (size_t)N * 4;
  float*    eal  = ws + o; o += (size_t)ET * 4;
  float*    pool = ws + o; o += (size_t)G * 64;
  float*    cnt  = ws + o; o += G;                 // contiguous after pool

  auto cdiv = [](long a, long b) { return (unsigned)((a + b - 1) / b); };

  auto run_layer = [&](const float* in, int Kin, const float* W,
                       const float* as_, const float* ad_, const float* b_,
                       int H, int C, int do_elu) {
    const int HC = H * C;                          // 256 or 64
    const int hcShift = (HC == 256) ? 8 : 6;
    const int cShift  = 6;                         // C == 64 always
    // 1) projection via WMMA (16x64 tile per wave)
    dim3 gg(cdiv(N, 16), HC / 64);
    if (Kin == 128)      gemm_xwt_wmma<128><<<gg, 32, 0, stream>>>(in, W, hbuf, N, HC);
    else if (Kin == 256) gemm_xwt_wmma<256><<<gg, 32, 0, stream>>>(in, W, hbuf, N, HC);
    else                 gemm_xwt_wmma<64> <<<gg, 32, 0, stream>>>(in, W, hbuf, N, HC);
    // 2) per-node attention logits
    node_alpha<<<cdiv((long)N * H, 256), 256, 0, stream>>>(hbuf, as_, ad_, asrc, adst, N, H, C);
    // 3) clear accumulators
    fill_u32<<<cdiv((long)N * H, 256), 256, 0, stream>>>(mkey, 0u, (long)N * H);
    fill_f32<<<cdiv((long)N * H, 256), 256, 0, stream>>>(den, 0.0f, (long)N * H);
    fill_f32<<<cdiv((long)N * HC, 256), 256, 0, stream>>>(agg, 0.0f, (long)N * HC);
    // 4) segment softmax over incoming edges
    edge_max <<<cdiv(ET, 256), 256, 0, stream>>>(srcI, dstI, E, N, asrc, adst, mkey, H);
    edge_exp <<<cdiv(ET, 256), 256, 0, stream>>>(srcI, dstI, E, N, asrc, adst, mkey, den, eal, H);
    edge_norm<<<cdiv((long)ET * H, 256), 256, 0, stream>>>(dstI, E, N, den, eal, H);
    // 5) weighted message aggregation
    edge_msg <<<cdiv((long)ET * HC, 256), 256, 0, stream>>>(srcI, dstI, E, N, hbuf, eal, agg,
                                                            H, hcShift, cShift);
    // 6) bias + optional ELU
    bias_act <<<cdiv((long)N * HC, 256), 256, 0, stream>>>(agg, b_, N, HC, do_elu);
  };

  // conv1: 128 -> 64 x 4 heads (concat), then ELU
  run_layer(x,   128, W1, a1s, a1d, b1, 4, 64, 1);
  // conv2: 256 -> 64
  run_layer(agg, 256, W2, a2s, a2d, b2, 1, 64, 0);
  // conv3: 64 -> 64
  run_layer(agg, 64,  W3, a3s, a3d, b3, 1, 64, 0);

  // global mean pool + linear head
  fill_f32<<<cdiv((long)G * 64 + G, 256), 256, 0, stream>>>(pool, 0.0f, (long)G * 64 + G);
  pool_sum<<<cdiv((long)N * 64, 256), 256, 0, stream>>>(agg, bat, pool, cnt, N, 64);
  head_fc<<<1, G * NCLS, 0, stream>>>(pool, cnt, Wfc, bfc, (float*)d_out, G, 64, NCLS);
}
